// deform_block_1726576853658
// MI455X (gfx1250) — matrix operand order repaired and checked by reading, not compiled, not run on hardware
//
#include <hip/hip_runtime.h>
#include <math.h>
#include <stdint.h>

// ---------------- problem constants ----------------
#define BB    4
#define CINC  64
#define COUTC 64
#define HH    192
#define WW    192
#define PH    194          // padded height (zero ring)
#define PW    194          // padded width
#define NTAP  9
#define KDIM  576          // 64 ch * 9 taps
#define HWSZ  (HH * WW)    // 36864
#define XSEG  6            // 192 / 32 column segments
#define NWG   (BB * HH * XSEG)  // 4608 workgroups for conv kernels

// ---------------- CDNA5 types ----------------
typedef __attribute__((ext_vector_type(16))) __bf16 v16bf;
typedef __attribute__((ext_vector_type(8)))  float  v8f;
typedef __attribute__((ext_vector_type(4)))  unsigned int u32x4;
typedef __attribute__((ext_vector_type(8)))  int i32x8;
typedef __attribute__((ext_vector_type(4)))  int i32x4;

struct alignas(16) U128 { unsigned int a, b, c, d; };
union FragU { U128 q[2]; v16bf v; };   // both members trivial -> union ok

__device__ __forceinline__ unsigned short f2bf(float f) {
  unsigned int u = __float_as_uint(f);
  u += 0x7fffu + ((u >> 16) & 1u);      // round-to-nearest-even
  return (unsigned short)(u >> 16);
}
__device__ __forceinline__ unsigned int pk2(float lo, float hi) {
  return ((unsigned int)f2bf(hi) << 16) | (unsigned int)f2bf(lo);
}

// ---------------- kernel 0a: weight packing + stats zero ----------------
// Bpm  : 32 rows x 576 (k = n*64 + c), rows 0-17 = w_p, 18-26 = w_m, 27-31 = 0
// Bconv: 64 rows x 576 (k = n*64 + c) from w_conv (o, c, ky, kx)
__global__ void prep_kernel(const float* __restrict__ wp,
                            const float* __restrict__ wm,
                            const float* __restrict__ wc,
                            unsigned short* __restrict__ Bpm,
                            unsigned short* __restrict__ Bconv,
                            float* __restrict__ stats) {
  int tid = threadIdx.x;
  for (int i = tid; i < 32 * KDIM; i += blockDim.x) {
    int o = i / KDIM, k = i % KDIM;
    int c = k & 63, n = k >> 6;
    float v = 0.f;
    if (o < 18)      v = wp[o * KDIM + c * 9 + n];
    else if (o < 27) v = wm[(o - 18) * KDIM + c * 9 + n];
    Bpm[i] = f2bf(v);
  }
  for (int i = tid; i < 64 * KDIM; i += blockDim.x) {
    int o = i / KDIM, k = i % KDIM;
    int c = k & 63, n = k >> 6;
    Bconv[i] = f2bf(wc[o * KDIM + c * 9 + n]);
  }
  for (int i = tid; i < 128; i += blockDim.x) stats[i] = 0.f;
}

// ---------------- kernel 0b: NCHW -> padded NHWC (f32 + bf16) ------------
// Padded image has a zero ring: padded (y,x) = original (y-1, x-1).
__global__ void pad_nhwc_kernel(const float* __restrict__ in,
                                float* __restrict__ outf,
                                unsigned short* __restrict__ outh) {
  int i = blockIdx.x * blockDim.x + threadIdx.x;
  if (i >= BB * PH * PW * CINC) return;
  int c = i & 63;
  int x = (i >> 6) % PW;
  int y = (i / (CINC * PW)) % PH;
  int b = i / (CINC * PW * PH);
  float v = 0.f;
  if (y >= 1 && y <= HH && x >= 1 && x <= WW)
    v = in[((b * CINC + c) * HH + (y - 1)) * WW + (x - 1)];
  outf[i] = v;
  outh[i] = f2bf(v);
}

// ---------------- TDM: load (c=64) x (x=34) x (y=3) bf16 tile to LDS -----
__device__ __forceinline__ void tdm_patch_load(const unsigned short* gsrc,
                                               void* lds_dst) {
  unsigned long long ga = (unsigned long long)(uintptr_t)gsrc;
  unsigned int lds = (unsigned int)(uintptr_t)lds_dst;   // low 32 bits = LDS offset

  u32x4 g0;
  g0[0] = 1u;                                        // count=1, user descriptor
  g0[1] = lds;                                       // lds_addr (bytes)
  g0[2] = (unsigned int)(ga & 0xffffffffu);          // global_addr[31:0]
  g0[3] = (unsigned int)((ga >> 32) & 0x1ffffffu)    // global_addr[56:32]
        | (2u << 30);                                // type = 2 ("image")

  i32x8 g1;
  g1[0] = 0x00010000;             // data_size = 1 (2-byte), mask=0, no pad/iter
  g1[1] = (64 << 16);             // tensor_dim0 = 64 (bits 79:48, low half)
  g1[2] = (PW << 16);             // tensor_dim0 hi = 0 | tensor_dim1 = 194
  g1[3] = (64 << 16);             // tensor_dim1 hi = 0 | tile_dim0 = 64
  g1[4] = (3 << 16) | 34;         // tile_dim1 = 34, tile_dim2 = 3
  g1[5] = 64;                     // tensor_dim0_stride = 64 (x step, elems)
  g1[6] = (int)((unsigned)(PW * 64) << 16);   // stride0 hi | stride1 lo16
  g1[7] = (int)((unsigned)(PW * 64) >> 16);   // tensor_dim1_stride hi bits

  i32x4 g2 = {PH, 0, 0, 0};       // tensor_dim2 = 194 (y length)
  i32x4 g3 = {0, 0, 0, 0};

#if __clang_major__ >= 23
  i32x8 z8 = {0, 0, 0, 0, 0, 0, 0, 0};
  __builtin_amdgcn_tensor_load_to_lds(g0, g1, g2, g3, z8, 0);
#else
  __builtin_amdgcn_tensor_load_to_lds(g0, g1, g2, g3, 0);
#endif
}

// ---------------- kernel 1: offset + modulation conv via WMMA ------------
// WG = 64 threads (2 waves), covers 32 pixels of one row (x0 .. x0+31).
// im2col GEMM: M=32 pixels (2 wave-tiles of 16), N=32 chans, K=576.
__global__ void __launch_bounds__(64)
conv_pm_kernel(const unsigned short* __restrict__ nhwcp_h,
               const unsigned short* __restrict__ Bpm,
               const float* __restrict__ bp,
               const float* __restrict__ bm,
               float* __restrict__ doff,
               float* __restrict__ dmod) {
  __shared__ alignas(16) unsigned short P[3 * 34 * 64];  // [ky][xx][c] bf16

  int bid  = blockIdx.x;
  int xseg = bid % XSEG;
  int y    = (bid / XSEG) % HH;
  int b    = bid / (XSEG * HH);
  int x0   = xseg * 32;
  int tid  = threadIdx.x;

  // TDM patch load: padded rows y..y+2, padded cols x0..x0+33, all 64 ch.
  // Issued by wave 0 only; TENSORcnt is per-wave so drain before the barrier.
  if (tid < 32) {
    const unsigned short* src = nhwcp_h + ((size_t)(b * PH + y) * PW + x0) * CINC;
    tdm_patch_load(src, (void*)P);
    __builtin_amdgcn_s_wait_tensorcnt(0);
  }
  __syncthreads();

  int lane = tid & 31;
  int wave = tid >> 5;
  int m    = lane & 15;
  int half = lane >> 4;
  int pix  = wave * 16 + m;       // A-row pixel for this lane
  int o0   = lane & 15;           // B-column within N-tile

  v8f acc0 = {0.f, 0.f, 0.f, 0.f, 0.f, 0.f, 0.f, 0.f};
  v8f acc1 = acc0;

  for (int s = 0; s < 18; ++s) {
    int ks = s * 32;
    FragU a;
    {
      int k0 = ks + half * 8;                 // K run 0 (8 elems)
      int n = k0 >> 6, c = k0 & 63;
      a.q[0] = *(const U128*)(&P[((n / 3) * 34 + (pix + n % 3)) * 64 + c]);
      int k1 = k0 + 16;                       // K run 1 (8 elems)
      n = k1 >> 6; c = k1 & 63;
      a.q[1] = *(const U128*)(&P[((n / 3) * 34 + (pix + n % 3)) * 64 + c]);
    }
    int ks2 = ks + half * 8;                  // B: lanes16-31 take K+8..15, K+24..31
    FragU fb0, fb1;
    fb0.q[0] = *(const U128*)(&Bpm[o0 * KDIM + ks2]);
    fb0.q[1] = *(const U128*)(&Bpm[o0 * KDIM + ks2 + 16]);
    fb1.q[0] = *(const U128*)(&Bpm[(o0 + 16) * KDIM + ks2]);
    fb1.q[1] = *(const U128*)(&Bpm[(o0 + 16) * KDIM + ks2 + 16]);
    acc0 = __builtin_amdgcn_wmma_f32_16x16x32_bf16(false, a.v, false, fb0.v,
                                                   (short)0, acc0, false, false);
    acc1 = __builtin_amdgcn_wmma_f32_16x16x32_bf16(false, a.v, false, fb1.v,
                                                   (short)0, acc1, false, false);
  }

  // epilogue: bias; sigmoid for modulation channels; scatter to NCHW maps
  #pragma unroll
  for (int nt = 0; nt < 2; ++nt) {
    v8f acc = nt ? acc1 : acc0;
    int o = nt * 16 + o0;
    #pragma unroll
    for (int j = 0; j < 8; ++j) {
      int xm = x0 + wave * 16 + half * 8 + j;      // D row M = j + 8*half
      float v = acc[j];
      if (o < 18) {
        doff[(b * 18 + o) * HWSZ + y * WW + xm] = v + bp[o];
      } else if (o < 27) {
        float t = v + bm[o - 18];
        dmod[(b * 9 + (o - 18)) * HWSZ + y * WW + xm] = 1.f / (1.f + __expf(-t));
      }
    }
  }
}

// ---------------- kernel 2: deformable sampling + main conv via WMMA -----
// WG = 64 threads (2 waves), 32 pixels of one row.
// Phase A: build A[32][576] bf16 in LDS (k = n*64 + c), bilinear * modulation.
// Phase B: per wave GEMM 16x64, K=576 -> 72 wmma_f32_16x16x32_bf16.
__global__ void __launch_bounds__(64)
deform_conv_kernel(const float* __restrict__ nhwcp,
                   const unsigned short* __restrict__ Bconv,
                   const float* __restrict__ doff,
                   const float* __restrict__ dmod,
                   float* __restrict__ convout,
                   float* __restrict__ stats) {
  __shared__ alignas(16) unsigned short As[32 * KDIM];   // 36864 B

  int bid  = blockIdx.x;
  int xseg = bid % XSEG;
  int y    = (bid / XSEG) % HH;
  int b    = bid / (XSEG * HH);
  int x0   = xseg * 32;
  int tid  = threadIdx.x;

  // ---- phase A: 32 pixels * 9 taps, 64 channels each (padded image) ----
  for (int u = tid; u < 32 * NTAP; u += 64) {
    int n   = u % NTAP;
    int pix = u / NTAP;
    int x   = x0 + pix;
    int pidx = y * WW + x;
    float dy = doff[(b * 18 + n) * HWSZ + pidx];
    float dx = doff[(b * 18 + 9 + n) * HWSZ + pidx];
    float md = dmod[(b * 9 + n) * HWSZ + pidx];

    const float HPm1 = (float)(PH - 1);   // 193, padded coord max
    float py = (float)(y + 1) + (float)(n / 3 - 1) + dy;
    float px = (float)(x + 1) + (float)(n % 3 - 1) + dx;
    float pyc = fminf(fmaxf(py, 0.f), HPm1);
    float pxc = fminf(fmaxf(px, 0.f), HPm1);
    float fy = floorf(py), fx = floorf(px);
    float qy0 = fminf(fmaxf(fy, 0.f), HPm1);
    float qy1 = fminf(fmaxf(fy + 1.f, 0.f), HPm1);
    float qx0 = fminf(fmaxf(fx, 0.f), HPm1);
    float qx1 = fminf(fmaxf(fx + 1.f, 0.f), HPm1);
    float wy0 = 1.f + (qy0 - pyc);
    float wy1 = 1.f - (qy1 - pyc);
    float wx0 = 1.f + (qx0 - pxc);
    float wx1 = 1.f - (qx1 - pxc);
    float g_lt = md * wy0 * wx0;
    float g_rb = md * wy1 * wx1;
    float g_lb = md * wy0 * wx1;
    float g_rt = md * wy1 * wx0;

    // padded-image corner pointers; zero ring supplies padding values
    int iy0 = (int)qy0, iy1 = (int)qy1;
    int ix0 = (int)qx0, ix1 = (int)qx1;
    const float* base = nhwcp + (size_t)b * PH * PW * CINC;
    const float* p_lt = base + ((size_t)iy0 * PW + ix0) * CINC;
    const float* p_rb = base + ((size_t)iy1 * PW + ix1) * CINC;
    const float* p_lb = base + ((size_t)iy0 * PW + ix1) * CINC;
    const float* p_rt = base + ((size_t)iy1 * PW + ix0) * CINC;

    unsigned short* dst = &As[pix * KDIM + n * 64];
    #pragma unroll 4
    for (int c4 = 0; c4 < 16; ++c4) {
      float4 lt = *(const float4*)(p_lt + c4 * 4);
      float4 rb = *(const float4*)(p_rb + c4 * 4);
      float4 lb = *(const float4*)(p_lb + c4 * 4);
      float4 rt = *(const float4*)(p_rt + c4 * 4);
      float r0 = g_lt * lt.x + g_rb * rb.x + g_lb * lb.x + g_rt * rt.x;
      float r1 = g_lt * lt.y + g_rb * rb.y + g_lb * lb.y + g_rt * rt.y;
      float r2 = g_lt * lt.z + g_rb * rb.z + g_lb * lb.z + g_rt * rt.z;
      float r3 = g_lt * lt.w + g_rb * rb.w + g_lb * lb.w + g_rt * rt.w;
      *(uint2*)(dst + c4 * 4) = make_uint2(pk2(r0, r1), pk2(r2, r3));
    }
  }
  __syncthreads();

  // ---- phase B: wave-level 16x64 GEMM, K = 576 ----
  int lane = tid & 31;
  int wave = tid >> 5;
  int m    = lane & 15;
  int half = lane >> 4;
  int pix  = wave * 16 + m;
  int o0   = lane & 15;

  v8f acc[4];
  v8f vz = {0.f, 0.f, 0.f, 0.f, 0.f, 0.f, 0.f, 0.f};
  #pragma unroll
  for (int nt = 0; nt < 4; ++nt) acc[nt] = vz;

  const unsigned short* arow = &As[pix * KDIM];
  for (int s = 0; s < 18; ++s) {
    int ks = s * 32;
    FragU a;
    int k0 = ks + half * 8;
    a.q[0] = *(const U128*)(arow + k0);
    a.q[1] = *(const U128*)(arow + k0 + 16);
    int ks2 = ks + half * 8;
    if (s + 1 < 18)
      __builtin_prefetch(Bconv + o0 * KDIM + ks2 + 32, 0, 1);
    #pragma unroll
    for (int nt = 0; nt < 4; ++nt) {
      const unsigned short* brow = Bconv + (nt * 16 + o0) * KDIM + ks2;
      FragU fb;
      fb.q[0] = *(const U128*)(brow);
      fb.q[1] = *(const U128*)(brow + 16);
      acc[nt] = __builtin_amdgcn_wmma_f32_16x16x32_bf16(false, a.v, false, fb.v,
                                                        (short)0, acc[nt], false, false);
    }
  }

  // ---- epilogue: NCHW store + BN partial sums ----
  #pragma unroll
  for (int nt = 0; nt < 4; ++nt) {
    int o = nt * 16 + o0;
    float* orow = convout + (size_t)(b * COUTC + o) * HWSZ + y * WW;
    float s = 0.f, s2 = 0.f;
    #pragma unroll
    for (int j = 0; j < 8; ++j) {
      float v = acc[nt][j];
      int xm = x0 + wave * 16 + half * 8 + j;
      orow[xm] = v;
      s += v;
      s2 += v * v;
    }
    atomicAdd(&stats[o], s);
    atomicAdd(&stats[64 + o], s2);
  }
}

// ---------------- kernel 3: BN (batch stats) + ReLU + MaxPool2 ----------
__global__ void bn_relu_pool_kernel(const float* __restrict__ convout,
                                    const float* __restrict__ stats,
                                    const float* __restrict__ gamma,
                                    const float* __restrict__ beta,
                                    float* __restrict__ out) {
  int i = blockIdx.x * blockDim.x + threadIdx.x;
  const int HO = HH / 2, WO = WW / 2;
  if (i >= BB * COUTC * HO * WO) return;
  int xo = i % WO;
  int yo = (i / WO) % HO;
  int o  = (i / (WO * HO)) % COUTC;
  int b  = i / (WO * HO * COUTC);
  float cnt  = (float)(BB * HWSZ);
  float mean = stats[o] / cnt;
  float var  = stats[64 + o] / cnt - mean * mean;
  float sc = gamma[o] * rsqrtf(var + 1e-5f);
  float sh = beta[o] - mean * sc;
  const float* base = convout + (size_t)(b * COUTC + o) * HWSZ + (2 * yo) * WW + 2 * xo;
  float v0 = fmaxf(base[0]      * sc + sh, 0.f);
  float v1 = fmaxf(base[1]      * sc + sh, 0.f);
  float v2 = fmaxf(base[WW]     * sc + sh, 0.f);
  float v3 = fmaxf(base[WW + 1] * sc + sh, 0.f);
  out[i] = fmaxf(fmaxf(v0, v1), fmaxf(v2, v3));
}

// ---------------- launch ----------------
extern "C" void kernel_launch(void* const* d_in, const int* in_sizes, int n_in,
                              void* d_out, int out_size, void* d_ws, size_t ws_size,
                              hipStream_t stream) {
  const float* inp   = (const float*)d_in[0];
  const float* w_p   = (const float*)d_in[1];
  const float* b_p   = (const float*)d_in[2];
  const float* w_m   = (const float*)d_in[3];
  const float* b_m   = (const float*)d_in[4];
  const float* w_c   = (const float*)d_in[5];
  const float* gamma = (const float*)d_in[6];
  const float* beta  = (const float*)d_in[7];
  float* out = (float*)d_out;

  char* ws = (char*)d_ws;
  float*          nhwcp   = (float*)(ws);                         // 38,539,264 B
  unsigned short* nhwcp_h = (unsigned short*)(ws + 38539264);     // 19,269,632 B
  float*          convout = (float*)(ws + 57808896);              // 37,748,736 B
  float*          doff    = (float*)(ws + 95557632);              // 10,616,832 B
  float*          dmod    = (float*)(ws + 106174464);             //  5,308,416 B
  unsigned short* Bpm     = (unsigned short*)(ws + 111482880);    //     36,864 B
  unsigned short* Bconv   = (unsigned short*)(ws + 111519744);    //     73,728 B
  float*          stats   = (float*)(ws + 111593472);             //        512 B

  prep_kernel<<<1, 256, 0, stream>>>(w_p, w_m, w_c, Bpm, Bconv, stats);
  pad_nhwc_kernel<<<(BB * PH * PW * CINC + 255) / 256, 256, 0, stream>>>(
      inp, nhwcp, nhwcp_h);
  conv_pm_kernel<<<NWG, 64, 0, stream>>>(nhwcp_h, Bpm, b_p, b_m, doff, dmod);
  deform_conv_kernel<<<NWG, 64, 0, stream>>>(nhwcp, Bconv, doff, dmod, convout, stats);
  bn_relu_pool_kernel<<<(BB * COUTC * 96 * 96 + 255) / 256, 256, 0, stream>>>(
      convout, stats, gamma, beta, out);
}
